// Wdmpnn_Conv_30107720745235
// MI455X (gfx1250) — compile-verified, matrix operand order checked by reading, not run on hardware
//
#include <hip/hip_runtime.h>

// ---------------------------------------------------------------------------
// D-MPNN conv for MI455X (gfx1250, wave32, WMMA).
// GEMMs: tall-skinny A (fp32, converted to bf16 in registers) x small B
// (pre-converted bf16, transposed, K-padded to mult-of-32, N-padded to 320).
// Each wave computes a 16x64 output strip (4 accumulators) -> A fragment
// reused across 4 v_wmma_f32_16x16x32_bf16 per k-step; grid.y=5 so the big
// A matrix (192MB, L2-sized) is streamed only 5x instead of 19x.
// B slab (64 cols x Kp) staged in LDS per block via
// global_load_async_to_lds_b128 (ASYNCcnt) and read with ds_load_b128.
// SpMM segment-sums: global_atomic_add_f32 (L2-resident target).
// ---------------------------------------------------------------------------

typedef __bf16 bf16_t;
typedef __attribute__((ext_vector_type(16))) __bf16 v16bf;
typedef __attribute__((ext_vector_type(8)))  __bf16 v8bf;
typedef __attribute__((ext_vector_type(8)))  float  v8f;

#define NND   20000
#define NED   160000
#define NNZD  960000
#define NDIM  133
#define EDIM  14
#define HID   300
#define NPP   320     // HID padded to 5 strips of 64 columns
#define KPMAX 448

// ---------------- utility kernels ----------------

__global__ void zero_f32(float* __restrict__ p, long n) {
  long i = (long)blockIdx.x * blockDim.x + threadIdx.x;
  long stride = (long)gridDim.x * blockDim.x;
  for (; i < n; i += stride) p[i] = 0.0f;
}

// W (K x HID, row-major fp32) -> Wt (NPP x Kp, bf16, transposed, zero padded)
__global__ void convert_w(const float* __restrict__ W, bf16_t* __restrict__ Wt,
                          int K, int Kp) {
  long i = (long)blockIdx.x * blockDim.x + threadIdx.x;
  long total = (long)NPP * Kp;
  if (i >= total) return;
  int n = (int)(i / Kp);
  int k = (int)(i % Kp);
  float v = (n < HID && k < K) ? W[(long)k * HID + n] : 0.0f;
  Wt[i] = (bf16_t)v;
}

// Xcat[e][k] = k<133 ? nodes[src[e]][k] : k<147 ? edge_attr[e][k-133] : 0
__global__ void build_xcat(const float* __restrict__ nodes,
                           const float* __restrict__ eattr,
                           const int*   __restrict__ esrc,
                           float* __restrict__ X) {
  long i = (long)blockIdx.x * blockDim.x + threadIdx.x;
  long total = (long)NED * 160;
  if (i >= total) return;
  int e = (int)(i / 160);
  int k = (int)(i % 160);
  float v = 0.0f;
  if (k < NDIM)              v = nodes[(long)esrc[e] * NDIM + k];
  else if (k < NDIM + EDIM)  v = eattr[(long)e * EDIM + (k - NDIM)];
  X[i] = v;
}

// Xa[n][k] = k<133 ? nodes[n][k] : k<433 ? sum_inc[n][k-133] : 0
__global__ void build_xa(const float* __restrict__ nodes,
                         const float* __restrict__ suminc,
                         float* __restrict__ X) {
  long i = (long)blockIdx.x * blockDim.x + threadIdx.x;
  long total = (long)NND * 448;
  if (i >= total) return;
  int n = (int)(i / 448);
  int k = (int)(i % 448);
  float v = 0.0f;
  if (k < NDIM)             v = nodes[(long)n * NDIM + k];
  else if (k < NDIM + HID)  v = suminc[(long)n * HID + (k - NDIM)];
  X[i] = v;
}

// S[rows[z]] += H[cols[z]]  over 300-dim rows, float4 chunks + f32 atomics
__global__ void spmm_scatter(const int* __restrict__ rows,
                             const int* __restrict__ cols,
                             const float* __restrict__ H,
                             float* __restrict__ S, int nnz) {
  long i = (long)blockIdx.x * blockDim.x + threadIdx.x;
  long total = (long)nnz * (HID / 4);     // 75 chunks per nnz
  if (i >= total) return;
  int z = (int)(i / (HID / 4));
  int c = (int)(i % (HID / 4)) * 4;
  int r  = rows[z];
  int cl = cols[z];
  const float4 v = *(const float4*)(H + (long)cl * HID + c);
  float* dst = S + (long)r * HID + c;
  unsafeAtomicAdd(dst + 0, v.x);   // -> global_atomic_add_f32
  unsafeAtomicAdd(dst + 1, v.y);
  unsafeAtomicAdd(dst + 2, v.z);
  unsafeAtomicAdd(dst + 3, v.w);
}

// ---------------- WMMA GEMM ----------------
// out[m][n] = relu( (resid? resid[m][n]:0) + bias[n] + sum_k A[m][k]*W[k][n] )
// One wave -> 16 rows x 64 cols (4 WMMA tiles). 8 waves (128 rows) / block.
// B slab (64 x Kp bf16) async-staged into LDS once per block.
__global__ __launch_bounds__(256)
void gemm_relu_wmma(const float* __restrict__ A, int lda, int M, int K, int Kfull,
                    const bf16_t* __restrict__ Bt, int Kp,
                    const float* __restrict__ bias,
                    const float* __restrict__ resid, int ldr,
                    float* __restrict__ out, int ldo) {
  __shared__ __align__(16) bf16_t lbs[64 * KPMAX];

  const int lane = threadIdx.x & 31;
  const int wave = threadIdx.x >> 5;
  const int hf   = lane >> 4;      // selects K sub-block per 16x16x32 layout
  const int l16  = lane & 15;

  const int tileM = blockIdx.x * 8 + wave;
  const int nBase = blockIdx.y * 64;

  // ---- async fill: LDS <- 64 consecutive Bt rows (flat copy, 16B chunks) ----
  {
    const int chunks = Kp << 3;                       // 64*Kp/8
    const unsigned lbase = (unsigned)(unsigned long long)(lbs);
    const unsigned long long gbase =
        (unsigned long long)(const void*)(Bt + (long)nBase * Kp);
    for (int i = threadIdx.x; i < chunks; i += 256) {
      unsigned loff = lbase + ((unsigned)i << 4);
      unsigned long long ga = gbase + ((unsigned long long)i << 4);
      asm volatile("global_load_async_to_lds_b128 %0, %1, off"
                   :: "v"(loff), "v"(ga) : "memory");
    }
    asm volatile("s_wait_asynccnt 0x0" ::: "memory");
  }
  __syncthreads();

  long mrow = (long)tileM * 16 + l16;
  if (mrow >= M) mrow = M - 1;               // clamp loads; stores are guarded
  const float* ap = A + mrow * (long)lda;

  v8f acc0 = {}, acc1 = {}, acc2 = {}, acc3 = {};

  int kk = 0;
  for (; kk < Kfull; kk += 32) {
    const int kb0 = kk + hf * 8;
    const int kb1 = kb0 + 16;
    const float4 a0 = *(const float4*)(ap + kb0);
    const float4 a1 = *(const float4*)(ap + kb0 + 4);
    const float4 a2 = *(const float4*)(ap + kb1);
    const float4 a3 = *(const float4*)(ap + kb1 + 4);
    if (kk + 64 < Kp) __builtin_prefetch(ap + kb0 + 32, 0, 3);

    v16bf af;
    af[0]  = (bf16_t)a0.x; af[1]  = (bf16_t)a0.y; af[2]  = (bf16_t)a0.z; af[3]  = (bf16_t)a0.w;
    af[4]  = (bf16_t)a1.x; af[5]  = (bf16_t)a1.y; af[6]  = (bf16_t)a1.z; af[7]  = (bf16_t)a1.w;
    af[8]  = (bf16_t)a2.x; af[9]  = (bf16_t)a2.y; af[10] = (bf16_t)a2.z; af[11] = (bf16_t)a2.w;
    af[12] = (bf16_t)a3.x; af[13] = (bf16_t)a3.y; af[14] = (bf16_t)a3.z; af[15] = (bf16_t)a3.w;

#define B_FRAG(T)                                                              \
    __builtin_shufflevector(                                                   \
        *(const v8bf*)&lbs[((T)*16 + l16) * Kp + kb0],                         \
        *(const v8bf*)&lbs[((T)*16 + l16) * Kp + kb1],                         \
        0,1,2,3,4,5,6,7,8,9,10,11,12,13,14,15)

    acc0 = __builtin_amdgcn_wmma_f32_16x16x32_bf16(false, af, false, B_FRAG(0),
                                                   (short)0, acc0, false, false);
    acc1 = __builtin_amdgcn_wmma_f32_16x16x32_bf16(false, af, false, B_FRAG(1),
                                                   (short)0, acc1, false, false);
    acc2 = __builtin_amdgcn_wmma_f32_16x16x32_bf16(false, af, false, B_FRAG(2),
                                                   (short)0, acc2, false, false);
    acc3 = __builtin_amdgcn_wmma_f32_16x16x32_bf16(false, af, false, B_FRAG(3),
                                                   (short)0, acc3, false, false);
  }

  // tail: K not a multiple of 32 for all our GEMMs; Kp == Kfull + 32
  {
    const int kb0 = kk + hf * 8;
    const int kb1 = kb0 + 16;
    v16bf af;
#pragma unroll
    for (int j = 0; j < 8; ++j) {
      const int k0 = kb0 + j, k1 = kb1 + j;
      af[j]     = (bf16_t)((k0 < K) ? ap[k0] : 0.0f);
      af[j + 8] = (bf16_t)((k1 < K) ? ap[k1] : 0.0f);
    }
    acc0 = __builtin_amdgcn_wmma_f32_16x16x32_bf16(false, af, false, B_FRAG(0),
                                                   (short)0, acc0, false, false);
    acc1 = __builtin_amdgcn_wmma_f32_16x16x32_bf16(false, af, false, B_FRAG(1),
                                                   (short)0, acc1, false, false);
    acc2 = __builtin_amdgcn_wmma_f32_16x16x32_bf16(false, af, false, B_FRAG(2),
                                                   (short)0, acc2, false, false);
    acc3 = __builtin_amdgcn_wmma_f32_16x16x32_bf16(false, af, false, B_FRAG(3),
                                                   (short)0, acc3, false, false);
  }
#undef B_FRAG

  // C/D layout: lane -> N, VGPR r -> M = r + 8*(lane>=16)
  const long mbase = (long)tileM * 16 + hf * 8;
#pragma unroll
  for (int t = 0; t < 4; ++t) {
    const int ncol = nBase + t * 16 + l16;
    if (ncol >= HID) continue;
    const float bv = bias[ncol];
    const v8f accv = (t == 0) ? acc0 : (t == 1) ? acc1 : (t == 2) ? acc2 : acc3;
#pragma unroll
    for (int r = 0; r < 8; ++r) {
      const long m = mbase + r;
      if (m < M) {
        float v = accv[r] + bv;
        if (resid) v += resid[m * (long)ldr + ncol];
        out[m * (long)ldo + ncol] = fmaxf(v, 0.0f);
      }
    }
  }
}

// ---------------- host orchestration ----------------

extern "C" void kernel_launch(void* const* d_in, const int* in_sizes, int n_in,
                              void* d_out, int out_size, void* d_ws, size_t ws_size,
                              hipStream_t stream) {
  const float* nodes = (const float*)d_in[0];
  const float* eattr = (const float*)d_in[1];
  const int*   esrc  = (const int*)d_in[2];
  const int*   d2o_r = (const int*)d_in[3];
  const int*   d2o_c = (const int*)d_in[4];
  const int*   inc_r = (const int*)d_in[5];
  const int*   inc_c = (const int*)d_in[6];
  const float* W1  = (const float*)d_in[7];  const float* b1  = (const float*)d_in[8];
  const float* Wm1 = (const float*)d_in[9];  const float* bm1 = (const float*)d_in[10];
  const float* Wm2 = (const float*)d_in[11]; const float* bm2 = (const float*)d_in[12];
  const float* Wm3 = (const float*)d_in[13]; const float* bm3 = (const float*)d_in[14];
  const float* Wa  = (const float*)d_in[15]; const float* ba  = (const float*)d_in[16];
  float* out = (float*)d_out;

  char* base = (char*)d_ws;
  size_t off = 0;
  auto take = [&](size_t bytes) -> void* {
    void* p = base + off;
    off = (off + bytes + 255) & ~(size_t)255;
    return p;
  };

  float*  B0   = (float*)take((size_t)NED * HID * 4);   // 192 MB
  float*  B1   = (float*)take((size_t)NED * HID * 4);   // 192 MB
  float*  B2   = (float*)take((size_t)NED * HID * 4);   // 192 MB
  float*  SI   = (float*)take((size_t)NND * HID * 4);   // 24 MB
  bf16_t* W1t  = (bf16_t*)take((size_t)NPP * 160 * 2);
  bf16_t* Wm1t = (bf16_t*)take((size_t)NPP * 320 * 2);
  bf16_t* Wm2t = (bf16_t*)take((size_t)NPP * 320 * 2);
  bf16_t* Wm3t = (bf16_t*)take((size_t)NPP * 320 * 2);
  bf16_t* Wat  = (bf16_t*)take((size_t)NPP * 448 * 2);

  float* XCAT = B1;   // E x 160 f32, dead before B1 is first reused
  float* XA   = B0;   // N x 448 f32, built after B0's last read

  auto cw = [&](const float* W, bf16_t* Wt, int K, int Kp) {
    long tot = (long)NPP * Kp;
    convert_w<<<dim3((unsigned)((tot + 255) / 256)), dim3(256), 0, stream>>>(W, Wt, K, Kp);
  };
  auto zero = [&](float* p, long n) {
    zero_f32<<<dim3(4096), dim3(256), 0, stream>>>(p, n);
  };
  auto spmm = [&](const int* r, const int* c, const float* H, float* S, int nnz) {
    long tot = (long)nnz * (HID / 4);
    spmm_scatter<<<dim3((unsigned)((tot + 255) / 256)), dim3(256), 0, stream>>>(r, c, H, S, nnz);
  };
  auto gemm = [&](const float* A, int lda, int M, int K, int Kp, const bf16_t* Bt,
                  const float* bias, const float* resid, int ldr, float* o, int ldo) {
    int mt = (M + 15) / 16;
    dim3 grid((unsigned)((mt + 7) / 8), NPP / 64);
    gemm_relu_wmma<<<grid, dim3(256), 0, stream>>>(A, lda, M, K, K & ~31, Bt, Kp,
                                                   bias, resid, ldr, o, ldo);
  };

  // weights -> bf16 transposed padded
  cw(W1, W1t, 147, 160);
  cw(Wm1, Wm1t, 300, 320);
  cw(Wm2, Wm2t, 300, 320);
  cw(Wm3, Wm3t, 300, 320);
  cw(Wa, Wat, 433, 448);

  // h0 = relu(cat(nodes[src], edge_attr) @ W1 + b1)      -> B0
  {
    long tot = (long)NED * 160;
    build_xcat<<<dim3((unsigned)((tot + 255) / 256)), dim3(256), 0, stream>>>(
        nodes, eattr, esrc, XCAT);
  }
  gemm(XCAT, 160, NED, 147, 160, W1t, b1, nullptr, 0, B0, HID);

  // layer 1: S1 = A h0 (B0 -> B1); h1 = relu(h0 + S1@Wm1 + bm1) -> B2
  zero(B1, (long)NED * HID);
  spmm(d2o_r, d2o_c, B0, B1, NNZD);
  gemm(B1, HID, NED, 300, 320, Wm1t, bm1, B0, HID, B2, HID);

  // layer 2: S2 = A h1 (B2 -> B0); h2 -> B1
  zero(B0, (long)NED * HID);
  spmm(d2o_r, d2o_c, B2, B0, NNZD);
  gemm(B0, HID, NED, 300, 320, Wm2t, bm2, B2, HID, B1, HID);

  // layer 3: S3 = A h2 (B1 -> B0); h3 -> B2
  zero(B0, (long)NED * HID);
  spmm(d2o_r, d2o_c, B1, B0, NNZD);
  gemm(B0, HID, NED, 300, 320, Wm3t, bm3, B1, HID, B2, HID);

  // node aggregation: sum_inc = A_inc h3 (E rows -> N rows)
  zero(SI, (long)NND * HID);
  spmm(inc_r, inc_c, B2, SI, NED);

  // atom = relu(cat(nodes, sum_inc) @ Wa + ba) -> d_out
  {
    long tot = (long)NND * 448;
    build_xa<<<dim3((unsigned)((tot + 255) / 256)), dim3(256), 0, stream>>>(nodes, SI, XA);
  }
  gemm(XA, 448, NND, 433, 448, Wat, ba, nullptr, 0, out, HID);
}
